// MLSAE_52286931862187
// MI455X (gfx1250) — matrix operand order repaired and checked by reading, not compiled, not run on hardware
//
#include <hip/hip_runtime.h>
#include <hip/hip_bf16.h>

// ---------------- problem constants (from reference setup) ----------------
#define ROWS   1024      // L*B*P = 4*1*256
#define DIM    768       // D
#define NLAT   32768     // N
#define TOPK   32
#define AUXK   256
#define DEADS  1000
#define DEADT  1e-3f
#define EPSF   1e-5f

typedef __attribute__((ext_vector_type(16))) __bf16 v16bf;
typedef __attribute__((ext_vector_type(8)))  __bf16 v8bf;
typedef __attribute__((ext_vector_type(8)))  float  v8f;

static __device__ __forceinline__ unsigned short f32_to_bf16(float f) {
    unsigned int u = __float_as_uint(f);
    unsigned int lsb = (u >> 16) & 1u;
    u += 0x7fffu + lsb;                    // round to nearest even
    return (unsigned short)(u >> 16);
}

// ---------------- 0: zero counters ----------------
__global__ void mlsae_zero_kernel(int* __restrict__ nz, int* __restrict__ cnt) {
    int i = blockIdx.x * blockDim.x + threadIdx.x;
    if (i < NLAT) nz[i] = 0;
    if (i == 0) cnt[0] = 0;
}

// ---------------- 1: standardize rows, emit bf16 xhat ----------------
__global__ __launch_bounds__(256)
void mlsae_std_kernel(const float* __restrict__ x, const float* __restrict__ pre_bias,
                      unsigned short* __restrict__ xhat,
                      float* __restrict__ mu, float* __restrict__ sd) {
    __shared__ float s1[256];
    __shared__ float s2[256];
    const int row = blockIdx.x;
    const int t = threadIdx.x;
    const float* xr = x + (size_t)row * DIM;
    float a = 0.f, b = 0.f;
    for (int j = t; j < DIM; j += 256) { float v = xr[j]; a += v; b += v * v; }
    s1[t] = a; s2[t] = b;
    __syncthreads();
    for (int s = 128; s > 0; s >>= 1) {
        if (t < s) { s1[t] += s1[t + s]; s2[t] += s2[t + s]; }
        __syncthreads();
    }
    const float m   = s1[0] / (float)DIM;
    const float var = (s2[0] - (float)DIM * m * m) / (float)(DIM - 1);  // ddof=1
    const float st  = sqrtf(fmaxf(var, 0.f));
    if (t == 0) { mu[row] = m; sd[row] = st; }
    const float inv = 1.f / (st + EPSF);
    for (int j = t; j < DIM; j += 256) {
        float v = (xr[j] - m) * inv - pre_bias[j];   // fold (x - pre_bias) into xhat
        xhat[(size_t)row * DIM + j] = f32_to_bf16(v);
    }
}

// ---------------- 2: W_enc f32 -> bf16 ----------------
__global__ void mlsae_cvt_kernel(const float* __restrict__ w, unsigned short* __restrict__ wb, int n) {
    int i = blockIdx.x * blockDim.x + threadIdx.x;
    int stride = gridDim.x * blockDim.x;
    for (; i < n; i += stride) wb[i] = f32_to_bf16(w[i]);
}

// ---------------- 3: W_dec [D,N] -> W_decT [N,D] (tiled transpose) ----------------
__global__ void mlsae_tr_kernel(const float* __restrict__ wd, float* __restrict__ wdt) {
    __shared__ float tile[32][33];
    const int n0 = blockIdx.x * 32;
    const int d0 = blockIdx.y * 32;
    const int tx = threadIdx.x;   // 0..31
    const int ty = threadIdx.y;   // 0..7
    for (int r = ty; r < 32; r += 8)
        tile[r][tx] = wd[(size_t)(d0 + r) * NLAT + (n0 + tx)];
    __syncthreads();
    for (int r = ty; r < 32; r += 8)
        wdt[(size_t)(n0 + r) * DIM + (d0 + tx)] = tile[tx][r];
}

// ---------------- 4: encode GEMM, bf16 WMMA, f32 accum ----------------
// block = 256 threads (8 wave32 waves). Block tile: 32 (M) x 128 (N).
// Each wave owns two 16x16 C tiles (M split), reusing each B fragment for
// two WMMAs -> 2x B traffic amortization. K loop: 768/32 = 24 steps, 48 WMMA.
__global__ __launch_bounds__(256)
void mlsae_encode_gemm(const unsigned short* __restrict__ xhat,
                       const unsigned short* __restrict__ wenc,
                       float* __restrict__ lat) {
    __shared__ unsigned short sA[32 * DIM];          // 48 KB A tile, shared by all 8 waves
    const int m0    = blockIdx.y * 32;
    const int nBase = blockIdx.x * 128;

    // cooperative A-tile stage: 32 rows x 768 bf16 = 3072 uint4
    {
        const uint4* src = reinterpret_cast<const uint4*>(xhat + (size_t)m0 * DIM);
        uint4* dst = reinterpret_cast<uint4*>(sA);
        for (int i = threadIdx.x; i < 3072; i += 256) dst[i] = src[i];
    }
    __syncthreads();

    const int wave = threadIdx.x >> 5;
    const int lane = threadIdx.x & 31;
    const int half = lane >> 4;        // 0: lanes 0-15, 1: lanes 16-31
    const int ln16 = lane & 15;
    const int n    = nBase + wave * 16 + ln16;
    const unsigned short* brow = wenc + (size_t)n * DIM;

    v8f acc0 = {};
    v8f acc1 = {};
    for (int kk = 0; kk < DIM; kk += 32) {
        __builtin_prefetch(brow + kk + 256, 0, 0);   // global_prefetch_b8 on the B stream
        // B fragment: lane-half h holds contiguous K = kk + 16h .. kk + 16h + 15 of column n
        const v16bf b = *reinterpret_cast<const v16bf*>(brow + kk + half * 16);
        // A fragments, contiguous layout matching B: lane-half h holds K = kk+16h..+15
        const v8bf alo0 = *reinterpret_cast<const v8bf*>(&sA[ln16 * DIM + kk + half * 16]);
        const v8bf ahi0 = *reinterpret_cast<const v8bf*>(&sA[ln16 * DIM + kk + half * 16 + 8]);
        const v8bf alo1 = *reinterpret_cast<const v8bf*>(&sA[(16 + ln16) * DIM + kk + half * 16]);
        const v8bf ahi1 = *reinterpret_cast<const v8bf*>(&sA[(16 + ln16) * DIM + kk + half * 16 + 8]);
        v16bf a0, a1;
#pragma unroll
        for (int i = 0; i < 8; ++i) {
            a0[i] = alo0[i]; a0[8 + i] = ahi0[i];
            a1[i] = alo1[i]; a1[8 + i] = ahi1[i];
        }
        acc0 = __builtin_amdgcn_wmma_f32_16x16x32_bf16(false, a0, false, b, (short)0, acc0, false, false);
        acc1 = __builtin_amdgcn_wmma_f32_16x16x32_bf16(false, a1, false, b, (short)0, acc1, false, false);
    }

    // C layout: VGPR r -> M = r + 8*half, N = ln16
    float* outBase = lat + (size_t)m0 * NLAT + n;
#pragma unroll
    for (int r = 0; r < 8; ++r) {
        const int m = r + half * 8;
        outBase[(size_t)m * NLAT]        = acc0[r];
        outBase[(size_t)(16 + m) * NLAT] = acc1[r];
    }
}

// ---------------- 5/7: per-row top-K (row resident in LDS; mutation-free lexicographic select) ----------------
__global__ __launch_bounds__(256)
void mlsae_topk_kernel(const float* __restrict__ lat,
                       const unsigned char* __restrict__ mask, int useMask,
                       int Ksel,
                       float* __restrict__ outV, float* __restrict__ outI,
                       int* __restrict__ outIdx,
                       int* __restrict__ nz, int doHits) {
    __shared__ float sv[NLAT];           // 128 KB: whole latent row in LDS (320 KB/WGP available)
    __shared__ float rv[256];
    __shared__ int   ri[256];
    __shared__ float sPrevV;
    __shared__ int   sPrevI;

    const int row = blockIdx.x;
    const int t = threadIdx.x;
    const float* lr = lat + (size_t)row * NLAT;
    for (int i = t; i < NLAT; i += 256) {
        float v = lr[i];
        if (useMask) v *= (mask[i] ? 1.0f : 0.0f);
        sv[i] = v;
    }
    if (t == 0) { sPrevV = __builtin_inff(); sPrevI = -1; }
    __syncthreads();

    for (int j = 0; j < Ksel; ++j) {
        const float pv = sPrevV;
        const int   pi = sPrevI;
        float bv = -__builtin_inff();
        int   bi = 0x7fffffff;
        for (int i = t; i < NLAT; i += 256) {
            const float v = sv[i];
            // admissible iff key (v, -i) strictly below previously selected key
            const bool adm = (v < pv) || (v == pv && i > pi);
            if (adm && (v > bv || (v == bv && i < bi))) { bv = v; bi = i; }
        }
        rv[t] = bv; ri[t] = bi;
        __syncthreads();
        for (int s = 128; s > 0; s >>= 1) {
            if (t < s) {
                const float ov = rv[t + s]; const int oi = ri[t + s];
                if (ov > rv[t] || (ov == rv[t] && oi < ri[t])) { rv[t] = ov; ri[t] = oi; }
            }
            __syncthreads();
        }
        if (t == 0) {
            const float v = rv[0]; const int idx = ri[0];
            outV[(size_t)row * Ksel + j]   = fmaxf(v, 0.f);   // relu
            outI[(size_t)row * Ksel + j]   = (float)idx;
            outIdx[(size_t)row * Ksel + j] = idx;
            if (doHits && v > DEADT) atomicAdd(&nz[idx], 1);
            sPrevV = v; sPrevI = idx;
        }
        __syncthreads();
    }
}

// ---------------- 6: dead-latent bookkeeping ----------------
__global__ void mlsae_dead_kernel(const int* __restrict__ last_nonzero, const int* __restrict__ nz,
                                  unsigned char* __restrict__ mask, int* __restrict__ cnt) {
    const int i = blockIdx.x * blockDim.x + threadIdx.x;
    if (i < NLAT) {
        const int hit = (nz[i] > 0) ? 1 : 0;
        const int ln = last_nonzero[i] * (1 - hit) + 1;
        const unsigned char m = (ln >= DEADS) ? 1 : 0;
        mask[i] = m;
        if (m) atomicAdd(cnt, 1);
    }
}

__global__ void mlsae_deadwrite_kernel(const int* __restrict__ cnt, float* __restrict__ out) {
    out[0] = (float)cnt[0] / (float)NLAT;
}

// ---------------- 8/9: sparse gather decode (coalesced W_decT rows) ----------------
__global__ __launch_bounds__(256)
void mlsae_decode_kernel(const float* __restrict__ vals, const int* __restrict__ idx,
                         const float* __restrict__ wdt, const float* __restrict__ pre_bias,
                         const float* __restrict__ mu, const float* __restrict__ sd,
                         float* __restrict__ out, int Ksel, int unstd) {
    __shared__ float svals[AUXK];
    __shared__ int   sidx[AUXK];
    const int row = blockIdx.x;
    const int t = threadIdx.x;
    if (t < Ksel) {
        svals[t] = vals[(size_t)row * Ksel + t];
        sidx[t]  = idx[(size_t)row * Ksel + t];
    }
    __syncthreads();
    const float m = mu[row], st = sd[row];
    for (int d = t; d < DIM; d += 256) {
        float acc = pre_bias[d];
        for (int k = 0; k < Ksel; ++k)
            acc += svals[k] * wdt[(size_t)sidx[k] * DIM + d];
        if (unstd) acc = acc * st + m;
        out[(size_t)row * DIM + d] = acc;
    }
}

// ---------------- launcher ----------------
extern "C" void kernel_launch(void* const* d_in, const int* in_sizes, int n_in,
                              void* d_out, int out_size, void* d_ws, size_t ws_size,
                              hipStream_t stream) {
    (void)in_sizes; (void)n_in; (void)out_size; (void)ws_size;

    const float* inputs       = (const float*)d_in[0];   // [4,1,256,768]
    const float* W_enc        = (const float*)d_in[1];   // [32768,768]
    const float* W_dec        = (const float*)d_in[2];   // [768,32768]
    const float* pre_bias     = (const float*)d_in[3];   // [768]
    const int*   last_nonzero = (const int*)d_in[4];     // [32768]

    // output layout (floats, reference return order)
    float* out    = (float*)d_out;
    float* o_tv   = out;                                   // 1024*32
    float* o_ti   = out + (size_t)ROWS * TOPK;             // 1024*32
    float* o_rec  = o_ti + (size_t)ROWS * TOPK;            // 1024*768
    float* o_av   = o_rec + (size_t)ROWS * DIM;            // 1024*256
    float* o_ai   = o_av + (size_t)ROWS * AUXK;            // 1024*256
    float* o_aux  = o_ai + (size_t)ROWS * AUXK;            // 1024*768
    float* o_dead = o_aux + (size_t)ROWS * DIM;            // 1

    // workspace layout (all offsets 256B-aligned)
    char* ws = (char*)d_ws;
    constexpr size_t OFF_XHAT = 0;                                        // 1024*768 u16
    constexpr size_t OFF_MU   = OFF_XHAT + (size_t)ROWS * DIM * 2;        // 1024 f32
    constexpr size_t OFF_SD   = OFF_MU + 4096;                            // 1024 f32
    constexpr size_t OFF_WENC = OFF_SD + 4096;                            // 32768*768 u16
    constexpr size_t OFF_LAT  = OFF_WENC + (size_t)NLAT * DIM * 2;        // 1024*32768 f32
    constexpr size_t OFF_WDT  = OFF_LAT + (size_t)ROWS * NLAT * 4;        // 32768*768 f32
    constexpr size_t OFF_NZ   = OFF_WDT + (size_t)NLAT * DIM * 4;         // 32768 i32
    constexpr size_t OFF_MASK = OFF_NZ + (size_t)NLAT * 4;                // 32768 u8
    constexpr size_t OFF_TI   = OFF_MASK + (size_t)NLAT;                  // 1024*32 i32
    constexpr size_t OFF_AI   = OFF_TI + (size_t)ROWS * TOPK * 4;         // 1024*256 i32
    constexpr size_t OFF_CNT  = OFF_AI + (size_t)ROWS * AUXK * 4;         // 1 i32

    unsigned short* w_xhat = (unsigned short*)(ws + OFF_XHAT);
    float*          w_mu   = (float*)(ws + OFF_MU);
    float*          w_sd   = (float*)(ws + OFF_SD);
    unsigned short* w_wenc = (unsigned short*)(ws + OFF_WENC);
    float*          w_lat  = (float*)(ws + OFF_LAT);
    float*          w_wdt  = (float*)(ws + OFF_WDT);
    int*            w_nz   = (int*)(ws + OFF_NZ);
    unsigned char*  w_mask = (unsigned char*)(ws + OFF_MASK);
    int*            w_ti   = (int*)(ws + OFF_TI);
    int*            w_ai   = (int*)(ws + OFF_AI);
    int*            w_cnt  = (int*)(ws + OFF_CNT);

    // 0) zero counters
    mlsae_zero_kernel<<<(NLAT + 255) / 256, 256, 0, stream>>>(w_nz, w_cnt);
    // 1) standardize -> bf16 xhat, mu, sd
    mlsae_std_kernel<<<ROWS, 256, 0, stream>>>(inputs, pre_bias, w_xhat, w_mu, w_sd);
    // 2) W_enc -> bf16
    mlsae_cvt_kernel<<<8192, 256, 0, stream>>>(W_enc, w_wenc, NLAT * DIM);
    // 3) W_dec -> W_decT
    {
        dim3 tb(32, 8), tg(NLAT / 32, DIM / 32);
        mlsae_tr_kernel<<<tg, tb, 0, stream>>>(W_dec, w_wdt);
    }
    // 4) encode GEMM (WMMA bf16), 32x128 block tile
    {
        dim3 g(NLAT / 128, ROWS / 32);
        mlsae_encode_gemm<<<g, 256, 0, stream>>>(w_xhat, w_wenc, w_lat);
    }
    // 5) top-32 + hit scatter
    mlsae_topk_kernel<<<ROWS, 256, 0, stream>>>(w_lat, w_mask, 0, TOPK,
                                                o_tv, o_ti, w_ti, w_nz, 1);
    // 6) dead mask + fraction
    mlsae_dead_kernel<<<(NLAT + 255) / 256, 256, 0, stream>>>(last_nonzero, w_nz, w_mask, w_cnt);
    mlsae_deadwrite_kernel<<<1, 1, 0, stream>>>(w_cnt, o_dead);
    // 7) masked top-256 (auxk)
    mlsae_topk_kernel<<<ROWS, 256, 0, stream>>>(w_lat, w_mask, 1, AUXK,
                                                o_av, o_ai, w_ai, w_nz, 0);
    // 8) recons (unstandardize) and 9) auxk_recons
    mlsae_decode_kernel<<<ROWS, 256, 0, stream>>>(o_tv, w_ti, w_wdt, pre_bias,
                                                  w_mu, w_sd, o_rec, TOPK, 1);
    mlsae_decode_kernel<<<ROWS, 256, 0, stream>>>(o_av, w_ai, w_wdt, pre_bias,
                                                  w_mu, w_sd, o_aux, AUXK, 0);
}
